// LSTMWithAttention_72748156059949
// MI455X (gfx1250) — matrix operand order repaired and checked by reading, not compiled, not run on hardware
//
#include <hip/hip_runtime.h>
#include <cstdint>
#include <cstddef>

// ---------------------------------------------------------------------------
// LSTM-with-attention caption decoder for MI455X (gfx1250, wave32, WMMA).
// Heavy GEMMs run as bf16 WMMA (v_wmma_f32_16x16x32_bf16) with fp32 accum.
// Weights are converted to bf16 once per launch; fc_logits_w (30.7MB bf16)
// stays L2-resident across the 32 recurrent steps.
// Each wave computes TWO adjacent 16x16 N-tiles sharing one A fragment:
// per K-chunk the mix is 6 x b128 loads : 2 x wmma (vs 4:1 single-tile),
// and the two independent accumulators hide WMMA->WMMA RAW latency.
// ---------------------------------------------------------------------------

typedef __bf16 bf16;
typedef __attribute__((ext_vector_type(16))) __bf16 v16bf;
typedef __attribute__((ext_vector_type(8)))  float  v8f;

#define NB      64          // batch
#define DVOCAB  30000
#define DEMBED  256
#define DANN    512
#define DHID    512
#define TMAX    32
#define DGLOB   2048
#define DL      289         // 17*17
#define DXCAT   1280        // DEMBED + DANN + DHID
#define DGATES  2048        // 4*DHID

__device__ __forceinline__ float sigmf(float x) { return 1.0f / (1.0f + expf(-x)); }

// ---------------------------------------------------------------------------
// fp32 -> bf16 conversion (weights once per launch; activations per step)
// ---------------------------------------------------------------------------
__global__ void cvt_bf16_kernel(const float* __restrict__ src, bf16* __restrict__ dst, long n) {
  long i = (long)blockIdx.x * blockDim.x + threadIdx.x;
  if (i < n) dst[i] = (bf16)src[i];
}

// Wcat[j, 0:768]   = w_ih[j, :]
// Wcat[j, 768:1280]= w_hh[j, :]
// bias_cat         = b_ih + b_hh
__global__ void build_wcat_kernel(const float* __restrict__ w_ih, const float* __restrict__ w_hh,
                                  const float* __restrict__ b_ih, const float* __restrict__ b_hh,
                                  bf16* __restrict__ Wcat, float* __restrict__ bias_cat) {
  long idx = (long)blockIdx.x * blockDim.x + threadIdx.x;
  const long total = (long)DGATES * DXCAT;
  if (idx < total) {
    long j = idx / DXCAT;
    long k = idx - j * DXCAT;
    float v = (k < 768) ? w_ih[j * 768 + k] : w_hh[j * 512 + (k - 768)];
    Wcat[idx] = (bf16)v;
  }
  if (idx < DGATES) bias_cat[idx] = b_ih[idx] + b_hh[idx];
}

__global__ void init_tok_kernel(int* __restrict__ tok) {
  if (threadIdx.x < NB) tok[threadIdx.x] = 1;   // START_TOKEN
}

// ---------------------------------------------------------------------------
// WMMA GEMM:  C[M=64, N] = A_bf16[64, K] @ W_bf16[N, K]^T + bias
// block = 128 threads (4 waves); wave w -> M-tile m0=16*w; blockIdx.x -> pair
// of N-tiles (n0, n0+16) sharing the A fragment. Tail pair (N not multiple of
// 32) computes a duplicated dead tile (wave-uniform predicate; EXEC all-1s).
// Optional fused strided secondary store (scatter logits into [B,V,T]).
// Fragment layouts follow CDNA5 ISA 7.12.2 (wave32).
// ---------------------------------------------------------------------------
__global__ void wmma_gemm_m64(const bf16* __restrict__ A, const bf16* __restrict__ W,
                              const float* __restrict__ bias, float* __restrict__ C,
                              float* __restrict__ C2, long c2_row_stride, long c2_col_stride,
                              int N, int K) {
  const int lane = threadIdx.x & 31;
  const int wave = threadIdx.x >> 5;
  const int n0 = blockIdx.x * 32;
  const int n1 = n0 + 16;
  const bool has2 = (n1 < N);               // wave-uniform
  const int m0 = wave * 16;

  // A fragment addressing (16x32 bf16): lane 0-15 -> K {0..7,16..23}; lane 16-31 -> K {8..15,24..31}
  const int m   = lane & 15;
  const int kbA = (lane < 16) ? 0 : 8;
  // B fragment addressing (32x16 bf16): lane&15 = N col; lanes 0-15 K {0..7,16..23}, lanes 16-31 K {8..15,24..31}
  const int nn  = lane & 15;
  const int kbB = (lane < 16) ? 0 : 8;

  const bf16* arow  = A + (size_t)(m0 + m) * K + kbA;
  const bf16* wrow0 = W + (size_t)(n0 + nn) * K + kbB;
  const bf16* wrow1 = W + (size_t)((has2 ? n1 : n0) + nn) * K + kbB;  // dup for tail

  v8f acc0 = {}, acc1 = {};
  union V16 { v16bf v; uint4 u[2]; };

  for (int k0 = 0; k0 < K; k0 += 32) {
    V16 a, b0, b1;
    a.u[0]  = *(const uint4*)(arow + k0);        // K = kbA+0 .. kbA+7
    a.u[1]  = *(const uint4*)(arow + k0 + 16);   // K = kbA+16 .. kbA+23
    b0.u[0] = *(const uint4*)(wrow0 + k0);
    b0.u[1] = *(const uint4*)(wrow0 + k0 + 16);
    b1.u[0] = *(const uint4*)(wrow1 + k0);
    b1.u[1] = *(const uint4*)(wrow1 + k0 + 16);
    // speculative stream prefetch of the weight rows (global_prefetch_b8)
    __builtin_prefetch(wrow0 + k0 + 256, 0, 1);
    __builtin_prefetch(wrow1 + k0 + 256, 0, 1);
    acc0 = __builtin_amdgcn_wmma_f32_16x16x32_bf16(false, a.v, false, b0.v,
                                                   (short)0, acc0, false, false);
    acc1 = __builtin_amdgcn_wmma_f32_16x16x32_bf16(false, a.v, false, b1.v,
                                                   (short)0, acc1, false, false);
  }

  const int col0  = n0 + nn;
  const float bv0 = bias ? bias[col0] : 0.0f;
  const int col1  = n1 + nn;
  const float bv1 = (bias && has2) ? bias[col1] : 0.0f;
  const int rbase = m0 + ((lane >> 4) << 3);    // C: VGPR i -> row m0+i (lanes 0-15) / m0+i+8 (16-31)
#pragma unroll
  for (int i = 0; i < 8; ++i) {
    const int row = rbase + i;
    const float v0 = acc0[i] + bv0;
    C[(size_t)row * N + col0] = v0;
    if (C2) C2[(size_t)row * c2_row_stride + (size_t)col0 * c2_col_stride] = v0;
    if (has2) {
      const float v1 = acc1[i] + bv1;
      C[(size_t)row * N + col1] = v1;
      if (C2) C2[(size_t)row * c2_row_stride + (size_t)col1 * c2_col_stride] = v1;
    }
  }
}

// ---------------------------------------------------------------------------
// Attention: per-sample block. scores = q . local[:, l]; softmax; ctx = attn . local
// local: [B, 512, 289] fp32.  Also writes attn map tile for step t.
// ---------------------------------------------------------------------------
__global__ void attn_kernel(const float* __restrict__ q, const float* __restrict__ local,
                            float* __restrict__ ctx, float* __restrict__ attn_out, int t) {
  const int b   = blockIdx.x;
  const int tid = threadIdx.x;                 // 256 threads
  __shared__ float sq[DANN];
  __shared__ float ss[DL];
  __shared__ float red[256];

  for (int i = tid; i < DANN; i += 256) sq[i] = q[b * DANN + i];
  __syncthreads();

  const float* lb = local + (size_t)b * DANN * DL;
  for (int l = tid; l < DL; l += 256) {
    float s = 0.0f;
    for (int a = 0; a < DANN; ++a) s += sq[a] * lb[(size_t)a * DL + l];
    ss[l] = s;
  }
  __syncthreads();

  // max reduce
  float mv = -3.402823466e+38f;
  for (int l = tid; l < DL; l += 256) mv = fmaxf(mv, ss[l]);
  red[tid] = mv; __syncthreads();
  for (int s = 128; s > 0; s >>= 1) { if (tid < s) red[tid] = fmaxf(red[tid], red[tid + s]); __syncthreads(); }
  const float mx = red[0];
  __syncthreads();

  // exp + sum reduce
  float lsum = 0.0f;
  for (int l = tid; l < DL; l += 256) { float e = expf(ss[l] - mx); ss[l] = e; lsum += e; }
  red[tid] = lsum; __syncthreads();
  for (int s = 128; s > 0; s >>= 1) { if (tid < s) red[tid] += red[tid + s]; __syncthreads(); }
  const float inv = 1.0f / red[0];
  __syncthreads();

  for (int l = tid; l < DL; l += 256) {
    float av = ss[l] * inv;
    ss[l] = av;
    attn_out[((size_t)b * TMAX + t) * DL + l] = av;
  }
  __syncthreads();

  for (int a = tid; a < DANN; a += 256) {
    const float* la = lb + (size_t)a * DL;
    float accv = 0.0f;
    for (int l = 0; l < DL; ++l) accv += ss[l] * la[l];
    ctx[b * DANN + a] = accv;
  }
}

// X = bf16([ embed_w[tok], ctx, h ])  -> [64, 1280]
__global__ void build_x_kernel(const int* __restrict__ tok, const float* __restrict__ embed_w,
                               const float* __restrict__ ctx, const float* __restrict__ h,
                               bf16* __restrict__ X) {
  int idx = blockIdx.x * blockDim.x + threadIdx.x;
  if (idx >= NB * DXCAT) return;
  int b = idx / DXCAT, k = idx - b * DXCAT;
  float v;
  if (k < DEMBED)             v = embed_w[(size_t)tok[b] * DEMBED + k];
  else if (k < DEMBED + DANN) v = ctx[b * DANN + (k - DEMBED)];
  else                        v = h[b * DHID + (k - DEMBED - DANN)];
  X[idx] = (bf16)v;
}

// LSTM pointwise: gates [64, 2048] split i|f|g|o -> update c, h; emit h in bf16 too.
__global__ void lstm_pw_kernel(const float* __restrict__ gates, float* __restrict__ c,
                               float* __restrict__ h, bf16* __restrict__ hb) {
  int idx = blockIdx.x * blockDim.x + threadIdx.x;
  if (idx >= NB * DHID) return;
  int b = idx >> 9, j = idx & 511;
  const float* g = gates + (size_t)b * DGATES;
  float ig = sigmf(g[j]);
  float fg = sigmf(g[j + 512]);
  float gg = tanhf(g[j + 1024]);
  float og = sigmf(g[j + 1536]);
  float c2 = fg * c[idx] + ig * gg;
  float h2 = og * tanhf(c2);
  c[idx] = c2; h[idx] = h2; hb[idx] = (bf16)h2;
}

// First-max argmax over 30000 logits per sample (matches jnp.argmax tie rule).
__global__ void argmax_kernel(const float* __restrict__ logits, int* __restrict__ tok,
                              float* __restrict__ captions, int t) {
  const int b   = blockIdx.x;
  const int tid = threadIdx.x;  // 256
  __shared__ float sv[256];
  __shared__ int   si[256];
  const float* lb = logits + (size_t)b * DVOCAB;
  float best = -3.402823466e+38f; int bi = 0;
  for (int v = tid; v < DVOCAB; v += 256) {
    float x = lb[v];
    if (x > best) { best = x; bi = v; }
  }
  sv[tid] = best; si[tid] = bi; __syncthreads();
  for (int s = 128; s > 0; s >>= 1) {
    if (tid < s) {
      if (sv[tid + s] > sv[tid] || (sv[tid + s] == sv[tid] && si[tid + s] < si[tid])) {
        sv[tid] = sv[tid + s]; si[tid] = si[tid + s];
      }
    }
    __syncthreads();
  }
  if (tid == 0) { tok[b] = si[0]; captions[b * TMAX + t] = (float)si[0]; }
}

// ---------------------------------------------------------------------------
extern "C" void kernel_launch(void* const* d_in, const int* in_sizes, int n_in,
                              void* d_out, int out_size, void* d_ws, size_t ws_size,
                              hipStream_t stream) {
  const float* local   = (const float*)d_in[0];
  const float* glob    = (const float*)d_in[1];
  const float* fc_h_w  = (const float*)d_in[2];
  const float* fc_h_b  = (const float*)d_in[3];
  const float* fc_c_w  = (const float*)d_in[4];
  const float* fc_c_b  = (const float*)d_in[5];
  const float* embed_w = (const float*)d_in[6];
  const float* attn_w  = (const float*)d_in[7];
  const float* attn_b  = (const float*)d_in[8];
  const float* w_ih    = (const float*)d_in[9];
  const float* b_ih    = (const float*)d_in[10];
  const float* w_hh    = (const float*)d_in[11];
  const float* b_hh    = (const float*)d_in[12];
  const float* fcl_w   = (const float*)d_in[13];
  const float* fcl_b   = (const float*)d_in[14];

  float* out          = (float*)d_out;
  float* out_captions = out;                                       // [64,32]
  float* out_logits   = out + (long)NB * TMAX;                     // [64,30000,32]
  float* out_attn     = out + (long)NB * TMAX + (long)NB * DVOCAB * TMAX; // [64,32,17,17]

  // ---- workspace carving (256B aligned slots) ----
  char* ws = (char*)d_ws;
  size_t off = 0;
  auto carve = [&](size_t bytes) -> char* {
    char* p = ws + off;
    off = (off + bytes + 255) & ~(size_t)255;
    return p;
  };
  bf16*  wcat_b   = (bf16*)carve((size_t)DGATES * DXCAT * 2);
  bf16*  logw_b   = (bf16*)carve((size_t)DVOCAB * DHID * 2);
  bf16*  attnw_b  = (bf16*)carve((size_t)DANN * DHID * 2);
  bf16*  fhw_b    = (bf16*)carve((size_t)DHID * DGLOB * 2);
  bf16*  fcw_b    = (bf16*)carve((size_t)DHID * DGLOB * 2);
  bf16*  glob_b   = (bf16*)carve((size_t)NB * DGLOB * 2);
  float* bias_cat = (float*)carve((size_t)DGATES * 4);
  float* h_f      = (float*)carve((size_t)NB * DHID * 4);
  float* c_f      = (float*)carve((size_t)NB * DHID * 4);
  bf16*  h_b      = (bf16*)carve((size_t)NB * DHID * 2);
  float* q_f      = (float*)carve((size_t)NB * DANN * 4);
  float* ctx_f    = (float*)carve((size_t)NB * DANN * 4);
  bf16*  x_b      = (bf16*)carve((size_t)NB * DXCAT * 2);
  float* gates_f  = (float*)carve((size_t)NB * DGATES * 4);
  float* logits_f = (float*)carve((size_t)NB * DVOCAB * 4);
  int*   tok_i    = (int*)carve((size_t)NB * 4);

  auto blocks = [](long n, int bs) { return (unsigned)((n + bs - 1) / bs); };
  auto ntiles = [](int N) { return (unsigned)((N + 31) / 32); };   // N-tile pairs

  // ---- one-time weight conversion (per launch; deterministic) ----
  cvt_bf16_kernel<<<blocks((long)DVOCAB * DHID, 256), 256, 0, stream>>>(fcl_w,  logw_b,  (long)DVOCAB * DHID);
  cvt_bf16_kernel<<<blocks((long)DANN * DHID, 256),   256, 0, stream>>>(attn_w, attnw_b, (long)DANN * DHID);
  cvt_bf16_kernel<<<blocks((long)DHID * DGLOB, 256),  256, 0, stream>>>(fc_h_w, fhw_b,   (long)DHID * DGLOB);
  cvt_bf16_kernel<<<blocks((long)DHID * DGLOB, 256),  256, 0, stream>>>(fc_c_w, fcw_b,   (long)DHID * DGLOB);
  cvt_bf16_kernel<<<blocks((long)NB * DGLOB, 256),    256, 0, stream>>>(glob,   glob_b,  (long)NB * DGLOB);
  build_wcat_kernel<<<blocks((long)DGATES * DXCAT, 256), 256, 0, stream>>>(w_ih, w_hh, b_ih, b_hh, wcat_b, bias_cat);
  init_tok_kernel<<<1, 64, 0, stream>>>(tok_i);

  // ---- initial hidden / cell state: h = glob @ fc_h_w.T + b ; c likewise ----
  wmma_gemm_m64<<<ntiles(DHID), 128, 0, stream>>>(glob_b, fhw_b, fc_h_b, h_f, nullptr, 0, 0, DHID, DGLOB);
  wmma_gemm_m64<<<ntiles(DHID), 128, 0, stream>>>(glob_b, fcw_b, fc_c_b, c_f, nullptr, 0, 0, DHID, DGLOB);
  cvt_bf16_kernel<<<blocks((long)NB * DHID, 256), 256, 0, stream>>>(h_f, h_b, (long)NB * DHID);

  // ---- 32 recurrent steps ----
  for (int t = 0; t < TMAX; ++t) {
    // q = h @ attn_w.T + attn_b            [64,512]
    wmma_gemm_m64<<<ntiles(DANN), 128, 0, stream>>>(h_b, attnw_b, attn_b, q_f, nullptr, 0, 0, DANN, DHID);
    // scores/softmax/ctx + attn-map output
    attn_kernel<<<NB, 256, 0, stream>>>(q_f, local, ctx_f, out_attn, t);
    // X = [embed(tok), ctx, h] in bf16     [64,1280]
    build_x_kernel<<<blocks((long)NB * DXCAT, 256), 256, 0, stream>>>(tok_i, embed_w, ctx_f, h_f, x_b);
    // gates = X @ Wcat.T + (b_ih + b_hh)   [64,2048]
    wmma_gemm_m64<<<ntiles(DGATES), 128, 0, stream>>>(x_b, wcat_b, bias_cat, gates_f, nullptr, 0, 0, DGATES, DXCAT);
    // LSTM cell update (writes h fp32 + bf16)
    lstm_pw_kernel<<<blocks((long)NB * DHID, 256), 256, 0, stream>>>(gates_f, c_f, h_f, h_b);
    // logits = h2 @ fc_logits_w.T + b      [64,30000]; fused scatter into out[B,V,T]
    wmma_gemm_m64<<<ntiles(DVOCAB), 128, 0, stream>>>(h_b, logw_b, fcl_b, logits_f,
                                                      out_logits + t, (long)DVOCAB * TMAX, (long)TMAX,
                                                      DVOCAB, DHID);
    // next token + caption output
    argmax_kernel<<<NB, 256, 0, stream>>>(logits_f, tok_i, out_captions, t);
  }
}
